// CustomFlaxViTSelfAttention_55147380081318
// MI455X (gfx1250) — matrix operand order repaired and checked by reading, not compiled, not run on hardware
//
#include <hip/hip_runtime.h>

typedef __attribute__((ext_vector_type(16))) __bf16 v16bf;
typedef __attribute__((ext_vector_type(8)))  float v8f;
typedef __attribute__((ext_vector_type(8)))  unsigned short us8;
typedef __attribute__((ext_vector_type(4)))  unsigned int  u32x4;

#define DEV __device__ __forceinline__

constexpr int B_ = 64, S_ = 577, D_ = 768, H_ = 12, HD_ = 64;
constexpr int SP_ = 608;                    // seq padded to multiple of 32
constexpr int M_ = B_ * S_;                 // 36928 = 577*64 (mult of 64)
constexpr int N_ = 3 * D_;                  // 2304 fused QKV columns

constexpr long long XBF_OFF = 0;
constexpr long long XBF_SZ  = (long long)M_ * D_ * 2;              // bf16 X
constexpr long long WT_OFF  = XBF_OFF + XBF_SZ;
constexpr long long WT_SZ   = (long long)N_ * D_ * 2;              // bf16 W^T [2304][768]
constexpr long long Q_OFF   = WT_OFF + WT_SZ;
constexpr long long QKV_SZ  = (long long)B_ * H_ * SP_ * HD_ * 2;  // per buffer
constexpr long long K_OFF   = Q_OFF + QKV_SZ;
constexpr long long V_OFF   = K_OFF + QKV_SZ;

DEV unsigned short f2bf(float f) {
  union { float f; unsigned int u; } x; x.f = f;
  unsigned int u = x.u;
  return (unsigned short)((u + 0x7FFFu + ((u >> 16) & 1u)) >> 16);
}

DEV v16bf make_frag(const unsigned short* p0, const unsigned short* p1) {
  union { v16bf v; us8 s[2]; } u;
  u.s[0] = *(const us8*)p0;   // global_load_b128
  u.s[1] = *(const us8*)p1;   // global_load_b128
  return u.v;
}

DEV v8f wmma_bf16(v16bf a, v16bf b, v8f c) {
  return __builtin_amdgcn_wmma_f32_16x16x32_bf16(false, a, false, b, (short)0, c, false, false);
}

// ---------------- zero pad region (Q,K,Vt contiguous) ----------------
__global__ void zero_ws(u32x4* __restrict__ p, int n16) {
  int i = blockIdx.x * 256 + threadIdx.x;
  if (i < n16) { u32x4 z = {0u,0u,0u,0u}; p[i] = z; }
}

// ---------------- fp32 -> bf16 convert of hidden_states ----------------
__global__ void conv_x(const float* __restrict__ x, unsigned short* __restrict__ xb, int n8) {
  int i = blockIdx.x * 256 + threadIdx.x;
  if (i >= n8) return;
  const float4* p = (const float4*)x + (size_t)i * 2;
  float4 a = p[0], b = p[1];
  us8 o;
  o[0]=f2bf(a.x); o[1]=f2bf(a.y); o[2]=f2bf(a.z); o[3]=f2bf(a.w);
  o[4]=f2bf(b.x); o[5]=f2bf(b.y); o[6]=f2bf(b.z); o[7]=f2bf(b.w);
  ((us8*)xb)[i] = o;
}

// ---------------- W[k][n] -> Wt[n][k] bf16 (fused 3 matrices) ----------------
__global__ void conv_w(const float* __restrict__ wq, const float* __restrict__ wk,
                       const float* __restrict__ wv, unsigned short* __restrict__ wt, int n) {
  int tid = blockIdx.x * 256 + threadIdx.x;
  if (tid >= n) return;
  int k   = tid % 768;
  int col = (tid / 768) % 768;
  int sel = tid / (768 * 768);
  const float* w = (sel == 0) ? wq : ((sel == 1) ? wk : wv);
  wt[tid] = f2bf(w[(size_t)k * 768 + col]);
}

// ---------------- fused QKV GEMM: [36928x768] @ [768x2304] ----------------
// block = 8 waves: 64M x 128N tile; wave = 32x32 (2x2 WMMA).
// Epilogue: +bias, Q*=1/8, scatter to per-head Q/K [bh][SP][64], Vt [bh][64][SP].
__global__ void qkv_gemm(const unsigned short* __restrict__ xb,
                         const unsigned short* __restrict__ wt,
                         const float* __restrict__ bqp, const float* __restrict__ bkp,
                         const float* __restrict__ bvp,
                         unsigned short* __restrict__ qb, unsigned short* __restrict__ kb,
                         unsigned short* __restrict__ vtb) {
  const int lane = threadIdx.x & 31, wid = threadIdx.x >> 5;
  const int l15 = lane & 15, lh = lane >> 4;
  const int rowbase = blockIdx.y * 64 + (wid >> 2) * 32;
  const int colbase = blockIdx.x * 128 + (wid & 3) * 32;
  // A frag: lane l15 = row, lh picks K sub-octet {0..7 | 8..15} (+16 for e[8..15])
  const unsigned short* xr0 = xb + (size_t)(rowbase + l15) * 768 + lh * 8;
  const unsigned short* xr1 = xr0 + (size_t)16 * 768;
  // B frag: lane l15 = column, lh picks K sub-octet {0..7 | 8..15} (+16 for e[8..15])
  const unsigned short* wr0 = wt + (size_t)(colbase + l15) * 768 + lh * 8;
  const unsigned short* wr1 = wr0 + (size_t)16 * 768;
  v8f acc[2][2];
  #pragma unroll
  for (int i = 0; i < 2; i++)
    #pragma unroll
    for (int j = 0; j < 2; j++)
      #pragma unroll
      for (int e = 0; e < 8; e++) acc[i][j][e] = 0.f;

  for (int kk = 0; kk < 768; kk += 32) {
    v16bf a0 = make_frag(xr0 + kk, xr0 + kk + 16);
    v16bf a1 = make_frag(xr1 + kk, xr1 + kk + 16);
    v16bf b0 = make_frag(wr0 + kk, wr0 + kk + 16);
    v16bf b1 = make_frag(wr1 + kk, wr1 + kk + 16);
    acc[0][0] = wmma_bf16(a0, b0, acc[0][0]);
    acc[0][1] = wmma_bf16(a0, b1, acc[0][1]);
    acc[1][0] = wmma_bf16(a1, b0, acc[1][0]);
    acc[1][1] = wmma_bf16(a1, b1, acc[1][1]);
  }

  #pragma unroll
  for (int i = 0; i < 2; i++)
    #pragma unroll
    for (int j = 0; j < 2; j++) {
      int col = colbase + j * 16 + l15;
      int sel = col / 768, cl = col % 768;
      const float* bias = (sel == 0) ? bqp : ((sel == 1) ? bkp : bvp);
      float bsv = bias[cl];
      float scale = (sel == 0) ? 0.125f : 1.0f;   // q / sqrt(64)
      int h = cl >> 6, d = cl & 63;
      unsigned short* dst = (sel == 0) ? qb : ((sel == 1) ? kb : vtb);
      #pragma unroll
      for (int r8 = 0; r8 < 8; r8++) {
        int r = rowbase + i * 16 + lh * 8 + r8;   // C layout: lanes>=16 -> M=j+8
        int b = r / 577;
        int s = r - b * 577;
        float val = (acc[i][j][r8] + bsv) * scale;
        size_t idx = (sel == 2)
          ? ((size_t)(b * 12 + h) * 64 + d) * SP_ + s          // Vt[bh][d][s]
          : ((size_t)(b * 12 + h) * SP_ + s) * 64 + d;         // Q/K[bh][s][d]
        dst[idx] = f2bf(val);
      }
    }
}

// ---------------- fused streaming attention ----------------
// block: one (b,h), 8 waves x 16 queries = 128 queries. Stream keys in chunks of 32.
// S^T tile = K-rows @ Q^T  ->  C(lane=query, vgpr=key) == bf16 A-frag layout for P@V.
__global__ void attn_kernel(const unsigned short* __restrict__ qb,
                            const unsigned short* __restrict__ kbuf,
                            const unsigned short* __restrict__ vtb,
                            const unsigned char* __restrict__ hmask,
                            float* __restrict__ out) {
  const int bh = blockIdx.x;
  const int b = bh / 12, h = bh % 12;
  const int lane = threadIdx.x & 31, wid = threadIdx.x >> 5;
  const int l15 = lane & 15, lh = lane >> 4;
  const int qbase = blockIdx.y * 128 + wid * 16;
  if (qbase >= S_) return;                         // wave-uniform early out
  const bool mk = hmask[h] != 0;

  const unsigned short* Qh = qb   + (size_t)bh * SP_ * 64;
  const unsigned short* Kh = kbuf + (size_t)bh * SP_ * 64;
  const unsigned short* Vh = vtb  + (size_t)bh * 64 * SP_;

  // Q as B-fragments (lane = query column of S^T), two hd chunks
  const unsigned short* qr = Qh + (size_t)(qbase + l15) * 64 + lh * 8;
  v16bf fq0 = make_frag(qr,      qr + 16);
  v16bf fq1 = make_frag(qr + 32, qr + 48);

  v8f oacc[4];
  #pragma unroll
  for (int t = 0; t < 4; t++)
    #pragma unroll
    for (int e = 0; e < 8; e++) oacc[t][e] = 0.f;
  float den = 0.f;

  for (int kk = 0; kk < SP_; kk += 32) {
    v8f st0, st1;
    #pragma unroll
    for (int e = 0; e < 8; e++) { st0[e] = 0.f; st1[e] = 0.f; }
    {
      const unsigned short* kr = Kh + (size_t)(kk + l15) * 64 + lh * 8;
      st0 = wmma_bf16(make_frag(kr,      kr + 16), fq0, st0);
      st0 = wmma_bf16(make_frag(kr + 32, kr + 48), fq1, st0);
      kr += (size_t)16 * 64;
      st1 = wmma_bf16(make_frag(kr,      kr + 16), fq0, st1);
      st1 = wmma_bf16(make_frag(kr + 32, kr + 48), fq1, st1);
    }
    if (mk) {                                      // ReLU branch: relu + L1 den
      #pragma unroll
      for (int j = 0; j < 8; j++) {
        st0[j] = fmaxf(st0[j], 0.f);
        st1[j] = fmaxf(st1[j], 0.f);
        den += st0[j] + st1[j];
      }
    }
    // pack two 16-key S^T tiles into one K=32 A-fragment (pure per-lane)
    union { v16bf v; unsigned short u[16]; } ap;
    #pragma unroll
    for (int j = 0; j < 8; j++) { ap.u[j] = f2bf(st0[j]); ap.u[8 + j] = f2bf(st1[j]); }
    #pragma unroll
    for (int td = 0; td < 4; td++) {
      const unsigned short* vr = Vh + (size_t)(td * 16 + l15) * SP_ + kk + lh * 8;
      oacc[td] = wmma_bf16(ap.v, make_frag(vr, vr + 16), oacc[td]);
    }
  }

  float sc[8];
  if (mk) {
    den += __shfl_xor(den, 16, 32);                // combine lane-half key ranges
    float invd = 1.0f / (den + 1e-5f);
    #pragma unroll
    for (int j = 0; j < 8; j++) sc[j] = __shfl(invd, j + lh * 8, 32);
  } else {
    #pragma unroll
    for (int j = 0; j < 8; j++) sc[j] = 1.0f / 577.0f;
  }
  #pragma unroll
  for (int td = 0; td < 4; td++) {
    int d = h * 64 + td * 16 + l15;
    #pragma unroll
    for (int j = 0; j < 8; j++) {
      int srow = qbase + lh * 8 + j;
      if (srow < S_)
        out[((size_t)b * S_ + srow) * 768 + d] = oacc[td][j] * sc[j];
    }
  }
}

extern "C" void kernel_launch(void* const* d_in, const int* in_sizes, int n_in,
                              void* d_out, int out_size, void* d_ws, size_t ws_size,
                              hipStream_t stream) {
  const float* hs = (const float*)d_in[0];
  const float* wq = (const float*)d_in[1];
  const float* bq = (const float*)d_in[2];
  const float* wk = (const float*)d_in[3];
  const float* bk = (const float*)d_in[4];
  const float* wv = (const float*)d_in[5];
  const float* bv = (const float*)d_in[6];
  const unsigned char* hm = (const unsigned char*)d_in[7];  // jnp bool_ = 1 byte

  char* ws = (char*)d_ws;
  unsigned short* xb   = (unsigned short*)(ws + XBF_OFF);
  unsigned short* wt   = (unsigned short*)(ws + WT_OFF);
  unsigned short* qbuf = (unsigned short*)(ws + Q_OFF);
  unsigned short* kbuf = (unsigned short*)(ws + K_OFF);
  unsigned short* vtb  = (unsigned short*)(ws + V_OFF);
  float* out = (float*)d_out;

  { int n16 = (int)(3 * QKV_SZ / 16);
    zero_ws<<<(n16 + 255) / 256, 256, 0, stream>>>((u32x4*)qbuf, n16); }
  { int n8 = M_ * D_ / 8;
    conv_x<<<(n8 + 255) / 256, 256, 0, stream>>>(hs, xb, n8); }
  { int n = 3 * 768 * 768;
    conv_w<<<(n + 255) / 256, 256, 0, stream>>>(wq, wk, wv, wt, n); }
  qkv_gemm<<<dim3(N_ / 128, M_ / 64), 256, 0, stream>>>(xb, wt, bq, bk, bv, qbuf, kbuf, vtb);
  attn_kernel<<<dim3(B_ * H_, (S_ + 127) / 128), 256, 0, stream>>>(qbuf, kbuf, vtb, hm, out);
}
